// MultiHeadAttention_49718541418949
// MI455X (gfx1250) — matrix operand order repaired and checked by reading, not compiled, not run on hardware
//
#include <hip/hip_runtime.h>

// MI455X / gfx1250 multi-head cross-attention.
// B=32, N=512, L=1024, D=512, H=8, C=64.  SCALE = 64^-0.5 = 0.125
//
// Pipeline (all f16 WMMA, f32 accumulate):
//   1) proj: Qh = x_q@Wq           -> (B,H,N,64) f16   [workspace]
//   2) proj: Kh = x_kv@Wk + rpb    -> (B,H,L,64) f16
//   3) proj: Vh = x_kv@Wv          -> (B,H,L,64) f16
//   4) attn: flash-style online softmax, double-buffered KV via async DMA,
//            O merged-head -> (B,N,512) f16
//   5) oproj: out = O@Wo + bo      -> (B,N,512) f32 (d_out)
//
// CDNA5 specifics: v_wmma_f32_16x16x32_f16 for all GEMMs,
// global_load_async_to_lds_b128 (ASYNCcnt) for pure-copy LDS fills,
// software-pipelined LDS double buffering, global_prefetch warming.

typedef __attribute__((ext_vector_type(16))) _Float16 v16h;
typedef __attribute__((ext_vector_type(8)))  _Float16 v8h;
typedef __attribute__((ext_vector_type(4)))  _Float16 v4h;
typedef __attribute__((ext_vector_type(8)))  float    v8f;

#define MHA_B 32
#define MHA_N 512
#define MHA_L 1024
#define MHA_D 512
#define MHA_H 8
#define MHA_C 64
#define MHA_SCALE 0.125f

// LDS byte offset of a __shared__ pointer: generic LDS addresses carry the
// LDS offset in addr[31:0] (aperture rule), so truncation recovers it.
__device__ __forceinline__ unsigned lds_off(const void* p) {
    return (unsigned)(uintptr_t)p;
}

// Async DMA 16 bytes global -> LDS, tracked on ASYNCcnt (no VGPR data path).
__device__ __forceinline__ void async_copy16(unsigned lds_addr, const void* gaddr) {
    asm volatile("global_load_async_to_lds_b128 %0, %1, off"
                 :: "v"(lds_addr), "v"((unsigned long long)(uintptr_t)gaddr)
                 : "memory");
}
__device__ __forceinline__ void wait_asynccnt0() {
    asm volatile("s_wait_asynccnt 0" ::: "memory");
}

// ---------------------------------------------------------------------------
// Projection GEMM: out[b,h,row,c] = f16( X[b,row,:] @ W[:, h*64+c] (+ rpb) )
// Block: 256 thr (8 waves). Tile: 128 rows x 64 cols (one head). K-step = 32.
// Each wave computes a 16x64 strip = 4 accumulators of 16x16.
// ---------------------------------------------------------------------------
template <bool RPB>
__global__ __launch_bounds__(256) void mha_proj_kernel(
    const float* __restrict__ X,    // (B, S, 512) f32
    const float* __restrict__ W,    // (512, 512) f32
    const float* __restrict__ rpb,  // (S, 512) f32 (used iff RPB)
    _Float16* __restrict__ out,     // (B, H, S, 64) f16
    int S)
{
    constexpr int PS = 40;  // padded halves per LDS row (32 data + 8), 80B: 16B aligned
    __shared__ alignas(16) _Float16 As[128 * PS];  // A tile: 128 rows x 32 k
    __shared__ alignas(16) _Float16 Bt[64 * PS];   // B tile transposed: Bt[n][k]

    const int b = blockIdx.z, h = blockIdx.y;
    const int row0 = blockIdx.x * 128;
    const int n0 = h * 64;
    const int tid = threadIdx.x, wave = tid >> 5, lane = tid & 31;
    const int m = lane & 15, hi = lane >> 4;

    const float* Xb = X + (size_t)b * S * MHA_D;

    v8f acc[4] = {};

    for (int k0 = 0; k0 < MHA_D; k0 += 32) {
        __syncthreads();
        // --- A tile: 128x32 f32 -> f16 (1024 float4 groups, 4/thread) ---
        #pragma unroll
        for (int e = 0; e < 4; ++e) {
            int idx = tid + e * 256;        // 8 float4-groups per row
            int r = idx >> 3, g = idx & 7;
            float4 f = ((const float4*)(Xb + (size_t)(row0 + r) * MHA_D + k0))[g];
            v4h st = {(_Float16)f.x, (_Float16)f.y, (_Float16)f.z, (_Float16)f.w};
            *(v4h*)&As[r * PS + g * 4] = st;
        }
        // --- B tile: W[k0+k][n0+n] -> Bt[n][k] (512 float4 groups, 2/thread) ---
        #pragma unroll
        for (int e = 0; e < 2; ++e) {
            int idx = tid + e * 256;        // k = idx>>4 (0..31), ngroup = idx&15
            int k = idx >> 4, ng = idx & 15;
            float4 f = ((const float4*)(W + (size_t)(k0 + k) * MHA_D + n0))[ng];
            int n = ng * 4;
            Bt[(n + 0) * PS + k] = (_Float16)f.x;
            Bt[(n + 1) * PS + k] = (_Float16)f.y;
            Bt[(n + 2) * PS + k] = (_Float16)f.z;
            Bt[(n + 3) * PS + k] = (_Float16)f.w;
        }
        __syncthreads();

        // A fragment: lane m = row, halves 0-7 -> k = hi*8+0..7, halves 8-15 -> k = 16+hi*8+0..7
        const _Float16* arow = &As[(wave * 16 + m) * PS];
        v16h af;
        *(v8h*)&af       = *(const v8h*)(arow + hi * 8);
        *((v8h*)&af + 1) = *(const v8h*)(arow + 16 + hi * 8);

        #pragma unroll
        for (int t = 0; t < 4; ++t) {
            // B fragment: lane m = column, halves 0-7 -> k = hi*8+0..7, halves 8-15 -> k = 16+hi*8+0..7
            const _Float16* brow = &Bt[(t * 16 + m) * PS];
            v16h bf;
            *(v8h*)&bf       = *(const v8h*)(brow + hi * 8);
            *((v8h*)&bf + 1) = *(const v8h*)(brow + 16 + hi * 8);
            acc[t] = __builtin_amdgcn_wmma_f32_16x16x32_f16(
                false, af, false, bf, (short)0, acc[t], false, false);
        }
    }

    // Optional rel-pos bias: batch all 32 loads, then adds (no per-elem waits).
    if constexpr (RPB) {
        float rv[4][8];
        #pragma unroll
        for (int t = 0; t < 4; ++t)
            #pragma unroll
            for (int r = 0; r < 8; ++r)
                rv[t][r] = rpb[(size_t)(row0 + wave * 16 + r + hi * 8) * MHA_D
                               + n0 + t * 16 + m];
        #pragma unroll
        for (int t = 0; t < 4; ++t)
            #pragma unroll
            for (int r = 0; r < 8; ++r)
                acc[t][r] += rv[t][r];
    }

    // Epilogue: C/D layout -> global f16 head-split. Row = r + hi*8, col = t*16 + m.
    #pragma unroll
    for (int t = 0; t < 4; ++t) {
        #pragma unroll
        for (int r = 0; r < 8; ++r) {
            int grow = row0 + wave * 16 + r + hi * 8;
            int gcol = t * 16 + m;
            out[(((size_t)b * MHA_H + h) * S + grow) * MHA_C + gcol] =
                (_Float16)acc[t][r];
        }
    }
}

// ---------------------------------------------------------------------------
// Flash attention per (b, h): block = 128 query rows, 8 waves x 16 rows each.
// KV streamed in 64-key tiles through LDS with DOUBLE BUFFERING: tile i+1's
// async K-DMA + V transpose are issued right after the barrier and overlap
// tile i's 16 WMMAs + softmax. Online softmax; O accumulated f32.
// ---------------------------------------------------------------------------
__global__ __launch_bounds__(256) void mha_attn_kernel(
    const _Float16* __restrict__ Qh,   // (B,H,N,64)
    const _Float16* __restrict__ Kh,   // (B,H,L,64)
    const _Float16* __restrict__ Vh,   // (B,H,L,64)
    const unsigned char* __restrict__ mask,  // (B,L) bool
    _Float16* __restrict__ Oh)         // (B,N,512) merged heads
{
    constexpr int PS = 72;  // 64 data + 8 pad halves; 144B rows, 16B aligned
    constexpr int NT = MHA_L / 64;                      // 16 kv tiles
    __shared__ alignas(16) _Float16 Kt[2][64 * PS];     // [buf][key][c]
    __shared__ alignas(16) _Float16 Vt[2][64 * PS];     // [buf][c][key] (transposed)
    __shared__ alignas(16) _Float16 Pb[8 * 16 * PS];    // per-wave P tile [16][64]

    const int b = blockIdx.z, h = blockIdx.y, qt = blockIdx.x;
    const int tid = threadIdx.x, wave = tid >> 5, lane = tid & 31;
    const int m = lane & 15, hi = lane >> 4;

    const size_t bh = (size_t)b * MHA_H + h;
    const _Float16* Qp = Qh + (bh * MHA_N + qt * 128) * MHA_C;
    const _Float16* Kp = Kh + bh * MHA_L * MHA_C;
    const _Float16* Vp = Vh + bh * MHA_L * MHA_C;
    const unsigned char* mb = mask + (size_t)b * MHA_L;

    // Per-thread slice for tile fills: 2 b128 chunks each.
    const int r0a = tid >> 3,         g0 = (tid & 7) * 8;          // chunk 0
    const int r1a = (tid + 256) >> 3, g1 = ((tid + 256) & 7) * 8;  // chunk 1

    // Q fragments stay resident: 16 rows x 64 c = two 16x32 A-frags.
    v16h qf[2];
    {
        const _Float16* qrow = Qp + (size_t)(wave * 16 + m) * MHA_C;
        #pragma unroll
        for (int cs = 0; cs < 2; ++cs) {
            *(v8h*)&qf[cs]       = *(const v8h*)(qrow + cs * 32 + hi * 8);
            *((v8h*)&qf[cs] + 1) = *(const v8h*)(qrow + cs * 32 + 16 + hi * 8);
        }
    }

    v8f acc[4] = {};
    float mrow[8], lrow[8];
    #pragma unroll
    for (int r = 0; r < 8; ++r) { mrow[r] = -1e30f; lrow[r] = 0.f; }

    _Float16* Pw = &Pb[wave * 16 * PS];

    // ---- prologue: fill buffer 0 with tile 0 ----
    {
        async_copy16(lds_off(&Kt[0][r0a * PS + g0]), Kp + (size_t)r0a * MHA_C + g0);
        async_copy16(lds_off(&Kt[0][r1a * PS + g1]), Kp + (size_t)r1a * MHA_C + g1);
        v8h v0 = *(const v8h*)(Vp + (size_t)r0a * MHA_C + g0);
        v8h v1 = *(const v8h*)(Vp + (size_t)r1a * MHA_C + g1);
        #pragma unroll
        for (int i = 0; i < 8; ++i) Vt[0][(g0 + i) * PS + r0a] = v0[i];
        #pragma unroll
        for (int i = 0; i < 8; ++i) Vt[0][(g1 + i) * PS + r1a] = v1[i];
    }

    for (int it = 0; it < NT; ++it) {
        const int kv0 = it * 64;
        const int cur = it & 1, nxt = cur ^ 1;

        wait_asynccnt0();      // my async stores into buf 'cur' done
        __syncthreads();       // + everyone's DS stores drained: tiles published

        // ---- issue tile it+1 into the other buffer (overlaps compute) ----
        if (it + 1 < NT) {
            const size_t base = (size_t)(kv0 + 64) * MHA_C;
            async_copy16(lds_off(&Kt[nxt][r0a * PS + g0]), Kp + base + (size_t)r0a * MHA_C + g0);
            async_copy16(lds_off(&Kt[nxt][r1a * PS + g1]), Kp + base + (size_t)r1a * MHA_C + g1);
            v8h v0 = *(const v8h*)(Vp + base + (size_t)r0a * MHA_C + g0);
            v8h v1 = *(const v8h*)(Vp + base + (size_t)r1a * MHA_C + g1);
            #pragma unroll
            for (int i = 0; i < 8; ++i) Vt[nxt][(g0 + i) * PS + r0a] = v0[i];
            #pragma unroll
            for (int i = 0; i < 8; ++i) Vt[nxt][(g1 + i) * PS + r1a] = v1[i];
        }
        // Warm L2 two tiles ahead.
        if (kv0 + 128 < MHA_L) {
            __builtin_prefetch(Kp + (size_t)(kv0 + 128) * MHA_C + tid * 16, 0, 3);
            __builtin_prefetch(Vp + (size_t)(kv0 + 128) * MHA_C + tid * 16, 0, 3);
        }

        // ---- S = Q K^T : 4 key-subtiles x 2 c-steps ----
        v8f sacc[4] = {};
        #pragma unroll
        for (int jt = 0; jt < 4; ++jt) {
            const _Float16* krow = &Kt[cur][(jt * 16 + m) * PS];
            #pragma unroll
            for (int cs = 0; cs < 2; ++cs) {
                v16h bf;
                *(v8h*)&bf       = *(const v8h*)(krow + cs * 32 + hi * 8);
                *((v8h*)&bf + 1) = *(const v8h*)(krow + cs * 32 + 16 + hi * 8);
                sacc[jt] = __builtin_amdgcn_wmma_f32_16x16x32_f16(
                    false, qf[cs], false, bf, (short)0, sacc[jt], false, false);
            }
        }

        // ---- scale + pad mask (column j = kv0 + jt*16 + m) ----
        float sreg[4][8];
        #pragma unroll
        for (int jt = 0; jt < 4; ++jt) {
            bool msk = mb[kv0 + jt * 16 + m] != 0;
            #pragma unroll
            for (int r = 0; r < 8; ++r) {
                float s = sacc[jt][r] * MHA_SCALE;
                sreg[jt][r] = msk ? -1e30f : s;
            }
        }

        // ---- online softmax (row = r + hi*8; xor-shuffles stay in half) ----
        float alpha[8];
        #pragma unroll
        for (int r = 0; r < 8; ++r) {
            float mx = fmaxf(fmaxf(sreg[0][r], sreg[1][r]),
                             fmaxf(sreg[2][r], sreg[3][r]));
            #pragma unroll
            for (int off = 8; off; off >>= 1) mx = fmaxf(mx, __shfl_xor(mx, off, 32));
            float mnew = fmaxf(mrow[r], mx);
            float a = __expf(mrow[r] - mnew);
            float sum = 0.f;
            #pragma unroll
            for (int jt = 0; jt < 4; ++jt) {
                float p = __expf(sreg[jt][r] - mnew);
                sreg[jt][r] = p;
                sum += p;
            }
            #pragma unroll
            for (int off = 8; off; off >>= 1) sum += __shfl_xor(sum, off, 32);
            lrow[r] = lrow[r] * a + sum;
            mrow[r] = mnew;
            alpha[r] = a;
        }

        // P -> per-wave LDS (C/D layout -> A layout round trip), rescale O acc
        #pragma unroll
        for (int jt = 0; jt < 4; ++jt)
            #pragma unroll
            for (int r = 0; r < 8; ++r)
                Pw[(r + hi * 8) * PS + jt * 16 + m] = (_Float16)sreg[jt][r];
        #pragma unroll
        for (int t = 0; t < 4; ++t)
            #pragma unroll
            for (int r = 0; r < 8; ++r)
                acc[t][r] *= alpha[r];

        // ---- O += P V : A-frags from Pw, B-frags from Vt ----
        #pragma unroll
        for (int ks = 0; ks < 2; ++ks) {
            const _Float16* prow = &Pw[m * PS + ks * 32];
            v16h pf;
            *(v8h*)&pf       = *(const v8h*)(prow + hi * 8);
            *((v8h*)&pf + 1) = *(const v8h*)(prow + 16 + hi * 8);
            #pragma unroll
            for (int t = 0; t < 4; ++t) {
                const _Float16* vrow = &Vt[cur][(t * 16 + m) * PS + ks * 32];
                v16h bf;
                *(v8h*)&bf       = *(const v8h*)(vrow + hi * 8);
                *((v8h*)&bf + 1) = *(const v8h*)(vrow + 16 + hi * 8);
                acc[t] = __builtin_amdgcn_wmma_f32_16x16x32_f16(
                    false, pf, false, bf, (short)0, acc[t], false, false);
            }
        }
    }

    // Epilogue: normalize, merged-head store (B,N,512)
    #pragma unroll
    for (int r = 0; r < 8; ++r) {
        float inv = 1.f / lrow[r];
        int grow = qt * 128 + wave * 16 + r + hi * 8;
        _Float16* orow = Oh + ((size_t)b * MHA_N + grow) * MHA_D + h * MHA_C;
        #pragma unroll
        for (int t = 0; t < 4; ++t)
            orow[t * 16 + m] = (_Float16)(acc[t][r] * inv);
    }
}

// ---------------------------------------------------------------------------
// Output projection: out(f32) = O(f16) @ Wo + bo.  A tile filled via async DMA.
// ---------------------------------------------------------------------------
__global__ __launch_bounds__(256) void mha_oproj_kernel(
    const _Float16* __restrict__ A,   // (B,N,512) f16
    const float* __restrict__ W,      // Wo (512,512)
    const float* __restrict__ bias,   // (512,)
    float* __restrict__ out)          // (B,N,512) f32
{
    constexpr int PS = 40;
    __shared__ alignas(16) _Float16 As[128 * PS];
    __shared__ alignas(16) _Float16 Bt[64 * PS];

    const int b = blockIdx.z, h = blockIdx.y;
    const int row0 = blockIdx.x * 128;
    const int n0 = h * 64;
    const int tid = threadIdx.x, wave = tid >> 5, lane = tid & 31;
    const int m = lane & 15, hi = lane >> 4;

    const _Float16* Ab = A + (size_t)b * MHA_N * MHA_D;

    v8f acc[4] = {};

    for (int k0 = 0; k0 < MHA_D; k0 += 32) {
        __syncthreads();
        // A tile: 128 rows x 32 halves, async DMA (512 x b128, 2/thread)
        #pragma unroll
        for (int e = 0; e < 2; ++e) {
            int idx = tid + e * 256;
            int r = idx >> 2, g = idx & 3;
            async_copy16(lds_off(&As[r * PS + g * 8]),
                         Ab + (size_t)(row0 + r) * MHA_D + k0 + g * 8);
        }
        // B tile transposed (f32 -> f16)
        #pragma unroll
        for (int e = 0; e < 2; ++e) {
            int idx = tid + e * 256;
            int k = idx >> 4, ng = idx & 15;
            float4 f = ((const float4*)(W + (size_t)(k0 + k) * MHA_D + n0))[ng];
            int n = ng * 4;
            Bt[(n + 0) * PS + k] = (_Float16)f.x;
            Bt[(n + 1) * PS + k] = (_Float16)f.y;
            Bt[(n + 2) * PS + k] = (_Float16)f.z;
            Bt[(n + 3) * PS + k] = (_Float16)f.w;
        }
        wait_asynccnt0();
        __syncthreads();

        const _Float16* arow = &As[(wave * 16 + m) * PS];
        v16h af;
        *(v8h*)&af       = *(const v8h*)(arow + hi * 8);
        *((v8h*)&af + 1) = *(const v8h*)(arow + 16 + hi * 8);
        #pragma unroll
        for (int t = 0; t < 4; ++t) {
            const _Float16* brow = &Bt[(t * 16 + m) * PS];
            v16h bf;
            *(v8h*)&bf       = *(const v8h*)(brow + hi * 8);
            *((v8h*)&bf + 1) = *(const v8h*)(brow + 16 + hi * 8);
            acc[t] = __builtin_amdgcn_wmma_f32_16x16x32_f16(
                false, af, false, bf, (short)0, acc[t], false, false);
        }
    }

    #pragma unroll
    for (int t = 0; t < 4; ++t) {
        int gcol = n0 + t * 16 + m;
        float bv = bias[gcol];
        #pragma unroll
        for (int r = 0; r < 8; ++r) {
            int grow = row0 + wave * 16 + r + hi * 8;
            out[((size_t)b * MHA_N + grow) * MHA_D + gcol] = acc[t][r] + bv;
        }
    }
}

// ---------------------------------------------------------------------------
extern "C" void kernel_launch(void* const* d_in, const int* in_sizes, int n_in,
                              void* d_out, int out_size, void* d_ws, size_t ws_size,
                              hipStream_t stream) {
    const float* x_q  = (const float*)d_in[0];
    const float* x_kv = (const float*)d_in[1];
    const unsigned char* pad = (const unsigned char*)d_in[2];
    const float* Wq  = (const float*)d_in[3];
    const float* Wk  = (const float*)d_in[4];
    const float* Wv  = (const float*)d_in[5];
    const float* Wo  = (const float*)d_in[6];
    const float* bo  = (const float*)d_in[7];
    const float* rpb = (const float*)d_in[8];

    char* ws = (char*)d_ws;
    const size_t szQ = (size_t)MHA_B * MHA_H * MHA_N * MHA_C * sizeof(_Float16); // 16 MB
    const size_t szK = (size_t)MHA_B * MHA_H * MHA_L * MHA_C * sizeof(_Float16); // 32 MB
    _Float16* Qh = (_Float16*)ws;
    _Float16* Kh = (_Float16*)(ws + szQ);
    _Float16* Vh = (_Float16*)(ws + szQ + szK);
    _Float16* Oh = (_Float16*)(ws + szQ + 2 * szK);  // (B,N,512) f16, 16 MB

    dim3 blk(256);
    mha_proj_kernel<false><<<dim3(MHA_N / 128, MHA_H, MHA_B), blk, 0, stream>>>(
        x_q, Wq, nullptr, Qh, MHA_N);
    mha_proj_kernel<true><<<dim3(MHA_L / 128, MHA_H, MHA_B), blk, 0, stream>>>(
        x_kv, Wk, rpb, Kh, MHA_L);
    mha_proj_kernel<false><<<dim3(MHA_L / 128, MHA_H, MHA_B), blk, 0, stream>>>(
        x_kv, Wv, nullptr, Vh, MHA_L);
    mha_attn_kernel<<<dim3(MHA_N / 128, MHA_H, MHA_B), blk, 0, stream>>>(
        Qh, Kh, Vh, pad, Oh);
    mha_oproj_kernel<<<dim3(MHA_N / 128, MHA_H, MHA_B), blk, 0, stream>>>(
        Oh, Wo, bo, (float*)d_out);
}
